// Encoder_3341484556346
// MI455X (gfx1250) — matrix operand order repaired and checked by reading, not compiled, not run on hardware
//
#include <hip/hip_runtime.h>

// ---------------------------------------------------------------------------
// LSTM encoder for gfx1250 (MI455X): bf16 WMMA (16x16x32) with f32 accum.
// Phase 1: fused embedding-gather + input GEMM  xz = bf16(emb[tok]) @ bf16(W) + b
// Phase 2: ONE persistent kernel (16 workgroups) runs all 256 timesteps,
//   separated by a grid-wide atomic barrier (release/acquire agent fences).
//   Per step:
//   - h staged into LDS via global_load_async_to_lds_b128 (ASYNCcnt path)
//   - each wave owns 2 column-tiles x 4 m-tiles -> U fragments read exactly once
//   - U fragments double-buffered 2 deep in registers to cover L2 latency
//   - gate strips combined through LDS (overlaid on the h buffer), sigmoid cell
//     update fused in-kernel.
// ---------------------------------------------------------------------------

typedef __attribute__((ext_vector_type(16))) __bf16 v16bf;
typedef __attribute__((ext_vector_type(8)))  __bf16 v8bf;
typedef __attribute__((ext_vector_type(8)))  float  v8f;

#define BATCH  64
#define SEQ    256
#define EMDIM  512
#define UNITS  1024
#define GATES  4096            // 4*UNITS, gate order [i,f,g,o]
#define MROWS  (BATCH*SEQ)     // 16384
#define ZSTR   260             // zbuf row stride (floats); 8*260 % 64 != 0
#define NBLK   16              // persistent scan grid size

union AFrag { v16bf v; v8bf h[2]; };

__device__ __forceinline__ v8f wmma_bf16(v16bf a, v16bf b, v8f c) {
  return __builtin_amdgcn_wmma_f32_16x16x32_bf16(false, a, false, b, (short)0, c,
                                                 false, false);
}

// A fragment (16x32 bf16) from a row-major bf16 matrix (lda = K stride).
__device__ __forceinline__ v16bf load_a_bf16(const __bf16* A, int lda, int m0,
                                             int k0, int lane) {
  int row = m0 + (lane & 15);
  int kh  = (lane >> 4) << 3;
  const __bf16* p = A + (size_t)row * lda + k0 + kh;
  AFrag a;
  a.h[0] = *(const v8bf*)(p);
  a.h[1] = *(const v8bf*)(p + 16);
  return a.v;
}

// A fragment from an f32 row (embedding gather), converted to bf16 on the fly.
__device__ __forceinline__ v16bf load_a_f32row(const float* rowp, int k0, int lane) {
  int kh = (lane >> 4) << 3;
  const float* p = rowp + k0 + kh;
  v8f x0 = *(const v8f*)(p);
  v8f x1 = *(const v8f*)(p + 16);
  AFrag a;
#pragma unroll
  for (int i = 0; i < 8; ++i) { a.v[i] = (__bf16)x0[i]; a.v[8 + i] = (__bf16)x1[i]; }
  return a.v;
}

// B fragment (32x16 bf16) from pre-transposed [N][K] bf16 matrix.
__device__ __forceinline__ v16bf load_b(const __bf16* Bt, int ldb, int n0,
                                        int k0, int lane) {
  int col = n0 + (lane & 15);
  int kh  = (lane >> 4) << 3;
  const __bf16* p = Bt + (size_t)col * ldb + k0 + kh;
  AFrag b;
  b.h[0] = *(const v8bf*)(p);
  b.h[1] = *(const v8bf*)(p + 16);
  return b.v;
}

__device__ __forceinline__ float sigmoidf(float x) {
  return 1.0f / (1.0f + __expf(-x));
}

// ---------------------------------------------------------------------------
// Prepasses
// ---------------------------------------------------------------------------

__global__ void init_hc_kernel(const float* __restrict__ h0,
                               const float* __restrict__ c0,
                               __bf16* __restrict__ hbf,
                               float* __restrict__ cws,
                               unsigned* __restrict__ barcnt, int n) {
  int i = blockIdx.x * blockDim.x + threadIdx.x;
  if (i == 0) *barcnt = 0u;                 // reset persistent-kernel barrier
  if (i < n) { hbf[i] = (__bf16)h0[i]; cws[i] = c0[i]; }
}

// src f32 [K][N] row-major -> dst bf16 [N][K] row-major (transpose + convert).
__global__ __launch_bounds__(256) void transpose_to_bf16_kernel(
    const float* __restrict__ src, __bf16* __restrict__ dst, int K, int N) {
  __shared__ float tile[32][33];
  int n0 = blockIdx.x * 32, k0 = blockIdx.y * 32;
  int tx = threadIdx.x, ty = threadIdx.y;
#pragma unroll
  for (int i = 0; i < 4; ++i)
    tile[ty + 8 * i][tx] = src[(size_t)(k0 + ty + 8 * i) * N + n0 + tx];
  __syncthreads();
#pragma unroll
  for (int i = 0; i < 4; ++i)
    dst[(size_t)(n0 + ty + 8 * i) * K + k0 + tx] = (__bf16)tile[tx][ty + 8 * i];
}

// ---------------------------------------------------------------------------
// Phase 1: xz[m][n] = sum_k bf16(emb[tokens[m]][k]) * bf16(W[k][n]) + b[n]
// ---------------------------------------------------------------------------
__global__ __launch_bounds__(256) void lstm_xz_gemm_kernel(
    const int* __restrict__ tokens, const float* __restrict__ emb,
    const __bf16* __restrict__ Wt,  // [GATES][EMDIM] bf16
    const float* __restrict__ bias, float* __restrict__ xz) {
  int lane = threadIdx.x & 31;
  int wave = threadIdx.x >> 5;
  int gw   = blockIdx.x * 8 + wave;
  int m0   = (gw >> 6) << 4;
  int nb   = (gw & 63) << 6;

  int tok = tokens[m0 + (lane & 15)];
  const float* erow = emb + (size_t)tok * EMDIM;

  v8f acc[4] = {v8f{}, v8f{}, v8f{}, v8f{}};
  for (int k0 = 0; k0 < EMDIM; k0 += 32) {
    v16bf a = load_a_f32row(erow, k0, lane);
#pragma unroll
    for (int q = 0; q < 4; ++q) {
      v16bf bf = load_b(Wt, EMDIM, nb + q * 16, k0, lane);
      acc[q] = wmma_bf16(a, bf, acc[q]);
    }
  }
  int mrow = m0 + ((lane >> 4) << 3);
#pragma unroll
  for (int q = 0; q < 4; ++q) {
    int n = nb + q * 16 + (lane & 15);
    float bv = bias[n];
#pragma unroll
    for (int r = 0; r < 8; ++r)
      xz[(size_t)(mrow + r) * GATES + n] = acc[q][r] + bv;
  }
}

// ---------------------------------------------------------------------------
// Phase 2: persistent scan over all SEQ timesteps.
// grid = 16 blocks (64 u-columns each) x 256 threads (8 waves).
// ---------------------------------------------------------------------------
__global__ __launch_bounds__(256) void lstm_scan_kernel(
    const float*  __restrict__ xz,     // [MROWS][GATES] (row = b*SEQ + t)
    const __bf16* __restrict__ Ut,     // [GATES][UNITS] bf16 (pre-transposed)
    __bf16*       __restrict__ hb0,    // ping h buffer (bf16(h0) at entry)
    __bf16*       __restrict__ hb1,    // pong h buffer
    float*        __restrict__ cstate, // [BATCH][UNITS] in/out
    float*        __restrict__ out,    // [BATCH][SEQ][UNITS]
    float*        __restrict__ out_h,  // [BATCH][UNITS]
    float*        __restrict__ out_c,  // [BATCH][UNITS]
    unsigned*     __restrict__ barcnt) {
  // 128 KB: h (bf16 [64][1024]) during the GEMM, overlaid by the f32 gate
  // staging buffer afterwards (barrier-separated).
  __shared__ __align__(16) char smem[BATCH * UNITS * sizeof(__bf16)];
  __bf16* hlds = (__bf16*)smem;
  float*  zbuf = (float*)smem;

  const int tid  = threadIdx.x;
  const int lane = tid & 31;
  const int w    = tid >> 5;
  const int ub   = blockIdx.x << 6;

  const int ct0 = w * 2, ct1 = ct0 + 1;
  const int ncol0 = ((ct0 >> 2) * UNITS) + ub + ((ct0 & 3) << 4);
  const int ncol1 = ((ct1 >> 2) * UNITS) + ub + ((ct1 & 3) << 4);

  for (int t = 0; t < SEQ; ++t) {
    const __bf16* hin  = (t & 1) ? hb1 : hb0;
    __bf16*       hout = (t & 1) ? hb0 : hb1;

    // ---- stage h into LDS with async copies: 32 x 16B per thread ----
#pragma unroll 4
    for (int i = 0; i < 32; ++i) {
      int off = (tid + i * 256) * 8;              // 8 bf16 = 16 bytes
      unsigned laddr = (unsigned)(size_t)(hlds + off);
      unsigned long long gaddr = (unsigned long long)(size_t)(hin + off);
      asm volatile("global_load_async_to_lds_b128 %0, %1, off"
                   :: "v"(laddr), "v"(gaddr) : "memory");
    }
    asm volatile("s_wait_asynccnt 0x0" ::: "memory");
    __syncthreads();

    v8f acc[2][4];
#pragma unroll
    for (int ci = 0; ci < 2; ++ci)
#pragma unroll
      for (int mt = 0; mt < 4; ++mt) acc[ci][mt] = v8f{};

    // ---- K loop, U fragments register-pipelined 2 deep ----
    v16bf b0a = load_b(Ut, UNITS, ncol0, 0,  lane);
    v16bf b0b = load_b(Ut, UNITS, ncol1, 0,  lane);
    v16bf b1a = load_b(Ut, UNITS, ncol0, 32, lane);
    v16bf b1b = load_b(Ut, UNITS, ncol1, 32, lane);
    for (int kk = 0; kk < 32; ++kk) {
      int k0 = kk << 5;
      v16bf bc0 = b0a, bc1 = b0b;
      b0a = b1a; b0b = b1b;
      if (kk < 30) {
        b1a = load_b(Ut, UNITS, ncol0, k0 + 64, lane);
        b1b = load_b(Ut, UNITS, ncol1, k0 + 64, lane);
      }
#pragma unroll
      for (int mt = 0; mt < 4; ++mt) {
        v16bf a = load_a_bf16(hlds, UNITS, mt << 4, k0, lane);  // ds path
        acc[0][mt] = wmma_bf16(a, bc0, acc[0][mt]);
        acc[1][mt] = wmma_bf16(a, bc1, acc[1][mt]);
      }
    }
    __syncthreads();   // all hlds reads done before zbuf overlays it

    // ---- epilogue: z = acc + xz_t -> LDS gate strips ----
    int mbase = (lane >> 4) << 3;
#pragma unroll
    for (int ci = 0; ci < 2; ++ci) {
      int ct = ct0 + ci;
      int gate = ct >> 2, nt = ct & 3;
      int nl = (nt << 4) + (lane & 15);
      int ng = gate * UNITS + ub + nl;
#pragma unroll
      for (int mt = 0; mt < 4; ++mt) {
#pragma unroll
        for (int r = 0; r < 8; ++r) {
          int m = (mt << 4) + mbase + r;
          float z = acc[ci][mt][r] + xz[((size_t)m * SEQ + t) * GATES + ng];
          zbuf[m * ZSTR + (gate << 6) + nl] = z;
        }
      }
    }
    __syncthreads();

    // ---- fused gate math: 4096 elems, 16 per thread ----
#pragma unroll
    for (int it = 0; it < 16; ++it) {
      int idx  = tid + it * 256;
      int rowm = idx >> 6;
      int col  = idx & 63;
      const float* zr = &zbuf[rowm * ZSTR + col];
      float i_ = sigmoidf(zr[0]);
      float f_ = sigmoidf(zr[64]);
      float g_ = sigmoidf(zr[128]);
      float o_ = sigmoidf(zr[192]);
      size_t ci = (size_t)rowm * UNITS + ub + col;
      float cn = f_ * cstate[ci] + i_ * g_;
      float hn = o_ * sigmoidf(cn);
      cstate[ci] = cn;
      out[((size_t)rowm * SEQ + t) * UNITS + ub + col] = hn;
      hout[ci] = (__bf16)hn;
      if (t == SEQ - 1) { out_h[ci] = hn; out_c[ci] = cn; }
    }

    // ---- grid-wide barrier between timesteps (skip after the last) ----
    if (t + 1 < SEQ) {
      __builtin_amdgcn_fence(__ATOMIC_RELEASE, "agent");  // flush h stores
      __syncthreads();
      if (tid == 0) {
        unsigned target = (unsigned)(t + 1) * NBLK;
        __hip_atomic_fetch_add(barcnt, 1u, __ATOMIC_RELAXED,
                               __HIP_MEMORY_SCOPE_AGENT);
        while (__hip_atomic_load(barcnt, __ATOMIC_RELAXED,
                                 __HIP_MEMORY_SCOPE_AGENT) < target)
          __builtin_amdgcn_s_sleep(1);
      }
      __syncthreads();
      __builtin_amdgcn_fence(__ATOMIC_ACQUIRE, "agent");  // invalidate stale h
    }
  }
}

// ---------------------------------------------------------------------------
// Host launcher
// ---------------------------------------------------------------------------
extern "C" void kernel_launch(void* const* d_in, const int* in_sizes, int n_in,
                              void* d_out, int out_size, void* d_ws, size_t ws_size,
                              hipStream_t stream) {
  const int*   tokens = (const int*)  d_in[0];
  const float* h0     = (const float*)d_in[1];
  const float* c0     = (const float*)d_in[2];
  const float* emb    = (const float*)d_in[3];
  const float* W      = (const float*)d_in[4];
  const float* U      = (const float*)d_in[5];
  const float* bias   = (const float*)d_in[6];
  float* out = (float*)d_out;

  char* ws = (char*)d_ws;
  size_t off = 0;
  float*  xz  = (float*) (ws + off); off += (size_t)MROWS * GATES * sizeof(float);
  __bf16* Ut  = (__bf16*)(ws + off); off += (size_t)GATES * UNITS * sizeof(__bf16);
  __bf16* Wt  = (__bf16*)(ws + off); off += (size_t)GATES * EMDIM * sizeof(__bf16);
  __bf16* hb0 = (__bf16*)(ws + off); off += (size_t)BATCH * UNITS * sizeof(__bf16);
  __bf16* hb1 = (__bf16*)(ws + off); off += (size_t)BATCH * UNITS * sizeof(__bf16);
  float*  cws = (float*) (ws + off); off += (size_t)BATCH * UNITS * sizeof(float);
  unsigned* barcnt = (unsigned*)(ws + off); off += 256;

  float* out_h = out + (size_t)BATCH * SEQ * UNITS;
  float* out_c = out_h + (size_t)BATCH * UNITS;

  init_hc_kernel<<<(BATCH * UNITS + 255) / 256, 256, 0, stream>>>(
      h0, c0, hb0, cws, barcnt, BATCH * UNITS);

  transpose_to_bf16_kernel<<<dim3(GATES / 32, EMDIM / 32), dim3(32, 8), 0, stream>>>(
      W, Wt, EMDIM, GATES);
  transpose_to_bf16_kernel<<<dim3(GATES / 32, UNITS / 32), dim3(32, 8), 0, stream>>>(
      U, Ut, UNITS, GATES);

  lstm_xz_gemm_kernel<<<8192, 256, 0, stream>>>(tokens, emb, Wt, bias, xz);

  lstm_scan_kernel<<<NBLK, 256, 0, stream>>>(xz, Ut, hb0, hb1, cws, out,
                                             out_h, out_c, barcnt);
}
